// CustomLSTM_84215718740626
// MI455X (gfx1250) — matrix operand order repaired and checked by reading, not compiled, not run on hardware
//
#include <hip/hip_runtime.h>
#include <hip/hip_bf16.h>
#include <cstdint>

typedef __attribute__((ext_vector_type(16))) __bf16 v16bf;
typedef __attribute__((ext_vector_type(8)))  __bf16 v8bf;
typedef __attribute__((ext_vector_type(8)))  float  v8f;
typedef __attribute__((ext_vector_type(4)))  unsigned int v4u;
typedef __attribute__((ext_vector_type(8)))  int v8i;
typedef __attribute__((ext_vector_type(4)))  int v4i;

#define HDIM   512
#define BATCH  32
#define TSTEPS 512
#define DDIM   512
#define NKI    3072   /* 6*H rows of Wi */
#define NKS    2560   /* 5*H rows of Ws */
#define NBLK_SCAN 32  /* persistent blocks in recurrence */

#if __has_builtin(__builtin_amdgcn_tensor_load_to_lds) && \
    __has_builtin(__builtin_amdgcn_s_wait_tensorcnt)
#define HAVE_TDM 1
#if __clang_major__ >= 23
#warning "CDNA5 probe: TDM path ENABLED (6-arg tensor_load_to_lds, clang>=23)"
#else
#warning "CDNA5 probe: TDM path ENABLED (5-arg tensor_load_to_lds, clang<=22)"
#endif
#else
#warning "CDNA5 probe: TDM builtin ABSENT - using vector-copy fallback"
#endif

__device__ __forceinline__ float fast_sigmoid(float x) {
    return 1.0f / (1.0f + __expf(-x));
}
__device__ __forceinline__ float fast_tanh(float x) {
    float e = __expf(2.0f * x);
    return (e - 1.0f) / (e + 1.0f);
}

__device__ __forceinline__ v16bf cat8(v8bf lo, v8bf hi) {
    return __builtin_shufflevector(lo, hi, 0,1,2,3,4,5,6,7,8,9,10,11,12,13,14,15);
}

// ---------------------------------------------------------------------------
// Pass 0: fp32 -> bf16 conversion of x, Wi, Ws; zero h double-buffer; reset
// the grid-barrier state (must be re-zeroed every call for graph replay).
// ---------------------------------------------------------------------------
__global__ __launch_bounds__(256)
void init_convert(const float* __restrict__ x,
                  const float* __restrict__ Wi,
                  const float* __restrict__ Ws,
                  __bf16* __restrict__ x_bf,
                  __bf16* __restrict__ Wi_bf,
                  __bf16* __restrict__ Ws_bf,
                  __bf16* __restrict__ h_buf,
                  int* __restrict__ bar) {
    long tid = (long)blockIdx.x * blockDim.x + threadIdx.x;
    long stride = (long)gridDim.x * blockDim.x;
    for (long i = tid; i < (long)BATCH * TSTEPS * DDIM; i += stride)
        x_bf[i] = (__bf16)x[i];
    for (long i = tid; i < (long)NKI * DDIM; i += stride)
        Wi_bf[i] = (__bf16)Wi[i];
    for (long i = tid; i < (long)NKS * HDIM; i += stride)
        Ws_bf[i] = (__bf16)Ws[i];
    for (long i = tid; i < 2 * BATCH * HDIM; i += stride)
        h_buf[i] = (__bf16)0.0f;
    if (tid < 8) bar[tid] = 0;
}

// ---------------------------------------------------------------------------
// Pass 1: pi = x @ Wi^T + bi   (M=B*T=16384, N=6H=3072, K=512)
// 32x32 tile per wave (2x2 WMMA register block): per K-chunk of 32 we issue
// 8 b128 loads and 4 v_wmma_f32_16x16x32_bf16. Output stored to (T, B, 6H).
// ---------------------------------------------------------------------------
__global__ __launch_bounds__(256)
void gemm_pi(const __bf16* __restrict__ x_bf,
             const __bf16* __restrict__ Wi_bf,
             const float*  __restrict__ bi,
             float* __restrict__ pi) {
    const int lane = threadIdx.x & 31;
    const int wid  = threadIdx.x >> 5;
    const int col  = lane & 15;
    const int half = lane >> 4;
    const int m0 = (blockIdx.y * 2 + (wid >> 2)) * 32;
    const int n0 = (blockIdx.x * 4 + (wid & 3)) * 32;

    v8f acc[2][2];
#pragma unroll
    for (int mi = 0; mi < 2; ++mi)
#pragma unroll
        for (int ni = 0; ni < 2; ++ni) acc[mi][ni] = (v8f){};

    const __bf16* aRow0 = x_bf + (size_t)(m0 + col) * DDIM;
    const __bf16* aRow1 = aRow0 + (size_t)16 * DDIM;
    const __bf16* bRow0 = Wi_bf + (size_t)(n0 + col) * DDIM;
    const __bf16* bRow1 = bRow0 + (size_t)16 * DDIM;

#pragma unroll 2
    for (int k0 = 0; k0 < DDIM; k0 += 32) {
        v16bf a0 = *(const v16bf*)(aRow0 + k0 + half * 16);
        v16bf a1 = *(const v16bf*)(aRow1 + k0 + half * 16);
        v16bf b0 = *(const v16bf*)(bRow0 + k0 + half * 16);
        v16bf b1 = *(const v16bf*)(bRow1 + k0 + half * 16);
        acc[0][0] = __builtin_amdgcn_wmma_f32_16x16x32_bf16(
            false, a0, false, b0, (short)0, acc[0][0], false, false);
        acc[0][1] = __builtin_amdgcn_wmma_f32_16x16x32_bf16(
            false, a0, false, b1, (short)0, acc[0][1], false, false);
        acc[1][0] = __builtin_amdgcn_wmma_f32_16x16x32_bf16(
            false, a1, false, b0, (short)0, acc[1][0], false, false);
        acc[1][1] = __builtin_amdgcn_wmma_f32_16x16x32_bf16(
            false, a1, false, b1, (short)0, acc[1][1], false, false);
    }

    float bias[2];
    bias[0] = bi[n0 + col];
    bias[1] = bi[n0 + 16 + col];
#pragma unroll
    for (int mi = 0; mi < 2; ++mi) {
#pragma unroll
        for (int r = 0; r < 8; ++r) {
            int mrow = m0 + mi * 16 + half * 8 + r;
            int bidx = mrow >> 9;     // / TSTEPS
            int t    = mrow & 511;    // % TSTEPS
            long base = ((long)t * BATCH + bidx) * NKI + n0 + col;
            pi[base]      = acc[mi][0][r] + bias[0];
            pi[base + 16] = acc[mi][1][r] + bias[1];
        }
    }
}

// ---------------------------------------------------------------------------
// Grid-wide sense barrier for the persistent scan kernel.
// ---------------------------------------------------------------------------
__device__ __forceinline__ void grid_barrier(int* cnt, int* gen, int nb, int& phase) {
    __syncthreads();
    __threadfence();
    if (threadIdx.x == 0) {
        if (__hip_atomic_fetch_add(cnt, 1, __ATOMIC_ACQ_REL, __HIP_MEMORY_SCOPE_AGENT) == nb - 1) {
            __hip_atomic_store(cnt, 0, __ATOMIC_RELAXED, __HIP_MEMORY_SCOPE_AGENT);
            __hip_atomic_fetch_add(gen, 1, __ATOMIC_RELEASE, __HIP_MEMORY_SCOPE_AGENT);
        } else {
            while (__hip_atomic_load(gen, __ATOMIC_ACQUIRE, __HIP_MEMORY_SCOPE_AGENT) < phase + 1)
                __builtin_amdgcn_s_sleep(2);
        }
    }
    phase += 1;
    __syncthreads();
}

// ---------------------------------------------------------------------------
// Pass 2: persistent recurrence. 32 blocks x 64 threads (2 waves).
// Block b owns h-columns [16b, 16b+16). Its 5 gate slices of Ws (5*16*512
// bf16 = 80 KB) are DMA'd into LDS by the Tensor Data Mover (one 2D-tile
// descriptor per gate, issued by wave 0; TDM ignores EXEC so the issuing
// branch is wave-uniform), then ds_load_b128-fed into WMMA for all 512 steps.
// c/h state lives in registers; h is exchanged via a double-buffered bf16
// global buffer with one grid barrier per step.
// ---------------------------------------------------------------------------
__global__ __launch_bounds__(64)
void lstm_scan(const float* __restrict__ pi,
               const __bf16* __restrict__ Ws_bf,
               const float* __restrict__ bs,
               const int*   __restrict__ lengths,
               __bf16* __restrict__ h_buf,
               float* __restrict__ ys,
               float* __restrict__ hT,
               float* __restrict__ cT,
               int* __restrict__ bar) {
    __shared__ __bf16 lws[5 * 16 * HDIM];   // 80 KB: this block's Ws slice

    const int j0   = blockIdx.x * 16;
    const int tid  = threadIdx.x;
    const int mh   = tid >> 5;        // batch half (wave id)
    const int lane = tid & 31;
    const int col  = lane & 15;
    const int half = lane >> 4;
    const int j    = j0 + col;

#ifdef HAVE_TDM
    // Tensor Data Mover: one 2D tile per gate (16 rows x 512 bf16, row
    // stride 512), global -> LDS. D# built per CDNA5 ISA 08 section 8.
    if (mh == 0) {
        unsigned lds_base = (unsigned)(uintptr_t)
            (__attribute__((address_space(3))) __bf16*)&lws[0];
        for (int q = 0; q < 5; ++q) {
            unsigned long long ga = (unsigned long long)(uintptr_t)
                (Ws_bf + (size_t)(q * HDIM + j0) * HDIM);
            v4u g0;
            g0.x = 1u;                                   // count=1 (valid)
            g0.y = lds_base + (unsigned)(q * 16 * HDIM * 2);  // lds_addr
            g0.z = (unsigned)(ga & 0xFFFFFFFFu);         // global_addr lo
            g0.w = (unsigned)((ga >> 32) & 0x1FFFFFFu)   // global_addr hi
                 | (2u << 30);                           // type=2 (image)
            v8i g1;
            g1[0] = (1 << 16);                 // data_size=1 (2B), mask=0
            g1[1] = (int)((unsigned)HDIM << 16);   // tensor_dim0[15:0] @48
            g1[2] = (int)(16u << 16);          // tensor_dim1[15:0] @80
            g1[3] = (int)((unsigned)HDIM << 16);   // tile_dim0 @112
            g1[4] = 16;                        // tile_dim1 @128
            g1[5] = HDIM;                      // tensor_dim0_stride lo @160
            g1[6] = 0;
            g1[7] = 0;
            v4i zz = {0, 0, 0, 0};
#if __clang_major__ >= 23
            v8i zz8 = {0, 0, 0, 0, 0, 0, 0, 0};
            __builtin_amdgcn_tensor_load_to_lds(g0, g1, zz, zz, zz8, 0);
#else
            __builtin_amdgcn_tensor_load_to_lds(g0, g1, zz, zz, 0);
#endif
        }
        __builtin_amdgcn_s_wait_tensorcnt(0);
    }
#else
    // Fallback: cooperative vector copy (rows q*H + [j0, j0+16) contiguous).
#pragma unroll
    for (int q = 0; q < 5; ++q) {
        const uint4* src = (const uint4*)(Ws_bf + (size_t)(q * HDIM + j0) * HDIM);
        uint4* dst = (uint4*)(lws + q * 16 * HDIM);
        for (int i = tid; i < 16 * HDIM / 8; i += 64) dst[i] = src[i];
    }
#endif
    __syncthreads();

    float bsv[5];
#pragma unroll
    for (int q = 0; q < 5; ++q) bsv[q] = bs[q * HDIM + j];

    int   len[8];
    float cpr[8], hpr[8];
#pragma unroll
    for (int r = 0; r < 8; ++r) {
        len[r] = lengths[mh * 16 + half * 8 + r];
        cpr[r] = 0.0f;
        hpr[r] = 0.0f;
    }

    const int batchA = mh * 16 + col;   // A-operand row for this lane
    int phase = 0;

    for (int t = 0; t < TSTEPS; ++t) {
        const __bf16* hin  = h_buf + (size_t)(t & 1) * (BATCH * HDIM);
        __bf16*       hout = h_buf + (size_t)((t + 1) & 1) * (BATCH * HDIM);

        v8f acc[5];
#pragma unroll
        for (int q = 0; q < 5; ++q) acc[q] = (v8f){};

        const __bf16* aRow = hin + (size_t)batchA * HDIM;
        __builtin_prefetch(pi + ((long)t * BATCH + batchA) * NKI + j, 0, 1);

        for (int k0 = 0; k0 < HDIM; k0 += 32) {
            v16bf a = *(const v16bf*)(aRow + k0 + half * 16);
#pragma unroll
            for (int q = 0; q < 5; ++q) {
                v16bf b = *(const v16bf*)(lws + (size_t)(q * 16 + col) * HDIM + k0 + half * 16);
                acc[q] = __builtin_amdgcn_wmma_f32_16x16x32_bf16(
                    false, a, false, b, (short)0, acc[q], false, false);
            }
        }

#pragma unroll
        for (int r = 0; r < 8; ++r) {
            const int bidx = mh * 16 + half * 8 + r;
            const long pbase = ((long)t * BATCH + bidx) * NKI + j;
            float pii = pi[pbase];
            float pif = pi[pbase + 1 * HDIM];
            float pig = pi[pbase + 2 * HDIM];
            float pio = pi[pbase + 3 * HDIM];
            float pir = pi[pbase + 4 * HDIM];
            float pih = pi[pbase + 5 * HDIM];

            float ig = fast_sigmoid(pii + acc[0][r] + bsv[0]);
            float fg = fast_sigmoid(pif + acc[1][r] + bsv[1]);
            float gg = fast_tanh  (pig + acc[2][r] + bsv[2]);
            float og = fast_sigmoid(pio + acc[3][r] + bsv[3]);
            float cn = ig * gg + fg * cpr[r];
            float out = og * fast_tanh(cn);
            float rg = fast_sigmoid(pir + acc[4][r] + bsv[4]);
            out = rg * out + (1.0f - rg) * pih;

            bool m = t < len[r];
            float hn = m ? out : hpr[r];
            cpr[r] = m ? cn : cpr[r];
            hpr[r] = hn;
            ys[((long)bidx * TSTEPS + t) * HDIM + j] = m ? out : 0.0f;
            hout[(size_t)bidx * HDIM + j] = (__bf16)hn;
        }

        grid_barrier(bar, bar + 1, NBLK_SCAN, phase);
    }

#pragma unroll
    for (int r = 0; r < 8; ++r) {
        const int bidx = mh * 16 + half * 8 + r;
        hT[(size_t)bidx * HDIM + j] = hpr[r];
        cT[(size_t)bidx * HDIM + j] = cpr[r];
    }
}

// ---------------------------------------------------------------------------
extern "C" void kernel_launch(void* const* d_in, const int* in_sizes, int n_in,
                              void* d_out, int out_size, void* d_ws, size_t ws_size,
                              hipStream_t stream) {
    (void)in_sizes; (void)n_in; (void)out_size; (void)ws_size;
    const float* x       = (const float*)d_in[0];
    const int*   lengths = (const int*)d_in[1];
    const float* Wi      = (const float*)d_in[2];
    const float* bi      = (const float*)d_in[3];
    const float* Ws      = (const float*)d_in[4];
    const float* bs      = (const float*)d_in[5];

    char* ws = (char*)d_ws;
    size_t off = 0;
    int*    bar   = (int*)(ws + off);    off += 256;
    __bf16* x_bf  = (__bf16*)(ws + off); off += (size_t)BATCH * TSTEPS * DDIM * 2;
    __bf16* Wi_bf = (__bf16*)(ws + off); off += (size_t)NKI * DDIM * 2;
    __bf16* Ws_bf = (__bf16*)(ws + off); off += (size_t)NKS * HDIM * 2;
    __bf16* h_buf = (__bf16*)(ws + off); off += (size_t)2 * BATCH * HDIM * 2;
    off = (off + 255) & ~(size_t)255;
    float*  pi    = (float*)(ws + off);  off += (size_t)BATCH * TSTEPS * NKI * 4;

    float* ys = (float*)d_out;
    float* hT = ys + (size_t)BATCH * TSTEPS * HDIM;
    float* cT = hT + (size_t)BATCH * HDIM;

    init_convert<<<2048, 256, 0, stream>>>(x, Wi, Ws, x_bf, Wi_bf, Ws_bf, h_buf, bar);
    gemm_pi<<<dim3(24, 256), 256, 0, stream>>>(x_bf, Wi_bf, bi, pi);
    lstm_scan<<<NBLK_SCAN, 64, 0, stream>>>(pi, Ws_bf, bs, lengths, h_buf,
                                            ys, hT, cT, bar);
}
